// FNetSublayer_69277822485277
// MI455X (gfx1250) — matrix operand order repaired and checked by reading, not compiled, not run on hardware
//
#include <hip/hip_runtime.h>

// ---------------------------------------------------------------------------
// FNet sublayer on MI455X (gfx1250): LayerNorm + 2D FFT(real) as f16 WMMA GEMMs.
//   Stage 1:  Yre = Xn @ CD,  Yim = Xn @ SD        (CD=cos, SD=-sin, D=1024)
//   Stage 2:  Out[b] = CS @ Yre[b] + PS @ Yim[b]   (CS=cos, PS=+sin, S=4096)
// Stage-1 output stored transposed (Y^T[b][d][s]) so both GEMMs consume
// B-operands in [n][k] layout.  Tile staging uses CDNA5 async global->LDS
// copies (ASYNCcnt) with double-buffered LDS when the builtins are available.
// ---------------------------------------------------------------------------

typedef __attribute__((ext_vector_type(16))) _Float16 v16h;
typedef __attribute__((ext_vector_type(8)))  float    v8f;
typedef __attribute__((ext_vector_type(4)))  int      v4i;

#define LDSS 40   // LDS tile row stride in halves (32 data + 8 pad -> conflict-free)

#if defined(__HIP_DEVICE_COMPILE__) && \
    __has_builtin(__builtin_amdgcn_global_load_async_to_lds_b128) && \
    __has_builtin(__builtin_amdgcn_s_wait_asynccnt)
#define USE_ASYNC 1
#else
#define USE_ASYNC 0
#endif

#if USE_ASYNC
typedef __attribute__((address_space(1))) v4i* gv4i_ptr;   // global (printed __device__)
typedef __attribute__((address_space(3))) v4i* lv4i_ptr;   // LDS    (printed __shared__)
__device__ __forceinline__ void async_b128(const void* g, void* l) {
    // DMA 16B from global directly into LDS; tracked by ASYNCcnt.
    __builtin_amdgcn_global_load_async_to_lds_b128(
        (gv4i_ptr)(unsigned long long)g,
        (lv4i_ptr)(unsigned int)(unsigned long long)l, 0, 0);
}
__device__ __forceinline__ void wait_async0() { __builtin_amdgcn_s_wait_asynccnt(0); }
#endif

union FragU { uint4 u[2]; v16h h; };
union PackU { uint4 u; _Float16 h[8]; };

// A fragment (16x32 f16): lanes 0-15 hold row m, K=0..7 (v0-3) and K=16..23
// (v4-7); lanes 16-31 hold K=8..15 and K=24..31.
__device__ __forceinline__ v16h frag_a(const _Float16* tile, int rowBase, int lane) {
    const int hi = (lane >> 4) << 3;                 // 0 or 8 halves
    const _Float16* p = tile + (rowBase + (lane & 15)) * LDSS + hi;
    FragU f;
    f.u[0] = *(const uint4*)(p);                     // K = hi .. hi+7
    f.u[1] = *(const uint4*)(p + 16);                // K = 16+hi .. 16+hi+7
    return f.h;
}

// B fragment (32x16 f16), tile stored as Bt[n][k]: lanes 0-15 hold col n with
// K=0..7 and K=16..23, lanes 16-31 hold K=8..15 and K=24..31 (matches frag_a).
__device__ __forceinline__ v16h frag_b(const _Float16* tile, int rowBase, int lane) {
    const _Float16* p = tile + (rowBase + (lane & 15)) * LDSS + ((lane >> 4) << 3);
    FragU f;
    f.u[0] = *(const uint4*)(p);
    f.u[1] = *(const uint4*)(p + 16);
    return f.h;
}

// ---------------------------------------------------------------------------
// DFT twiddle tables: C[idx]=cos(2*pi*r/N), S[idx]=ssign*sin(2*pi*r/N),
// r = (n*k) mod N (exact for power-of-two N -> good float precision).
// ---------------------------------------------------------------------------
__global__ void fill_dft(_Float16* __restrict__ C, _Float16* __restrict__ S,
                         int logN, float ssign) {
    const unsigned long long idx = (unsigned long long)blockIdx.x * 256ull + threadIdx.x;
    const unsigned N = 1u << logN;
    const unsigned n = (unsigned)(idx >> logN);
    const unsigned k = (unsigned)idx & (N - 1u);
    const unsigned r = (n * k) & (N - 1u);
    const float theta = (float)r * (6.2831853071795864769f / (float)N);
    float s, c;
    __sincosf(theta, &s, &c);
    C[idx] = (_Float16)c;
    S[idx] = (_Float16)(ssign * s);
}

// ---------------------------------------------------------------------------
// LayerNorm over D=1024, one block (8 waves) per row; writes f16 Xn.
// ---------------------------------------------------------------------------
__global__ __launch_bounds__(256) void ln_kernel(const float* __restrict__ x,
                                                 const float* __restrict__ gamma,
                                                 const float* __restrict__ beta,
                                                 _Float16* __restrict__ xn) {
    __shared__ float red[256];
    const int row = blockIdx.x;                     // 16384 rows
    const int t = threadIdx.x;
    const float* xr = x + (size_t)row * 1024;
    const float4 v = *(const float4*)(xr + t * 4);

    red[t] = v.x + v.y + v.z + v.w;
    __syncthreads();
    for (int off = 128; off > 0; off >>= 1) {
        if (t < off) red[t] += red[t + off];
        __syncthreads();
    }
    const float mean = red[0] * (1.0f / 1024.0f);
    __syncthreads();

    red[t] = v.x * v.x + v.y * v.y + v.z * v.z + v.w * v.w;
    __syncthreads();
    for (int off = 128; off > 0; off >>= 1) {
        if (t < off) red[t] += red[t + off];
        __syncthreads();
    }
    const float var = red[0] * (1.0f / 1024.0f) - mean * mean;
    const float rstd = rsqrtf(var + 1e-5f);

    const float4 g = *(const float4*)(gamma + t * 4);
    const float4 bb = *(const float4*)(beta + t * 4);
    union { _Float16 h[4]; uint2 u; } pk;
    pk.h[0] = (_Float16)((v.x - mean) * rstd * g.x + bb.x);
    pk.h[1] = (_Float16)((v.y - mean) * rstd * g.y + bb.y);
    pk.h[2] = (_Float16)((v.z - mean) * rstd * g.z + bb.z);
    pk.h[3] = (_Float16)((v.w - mean) * rstd * g.w + bb.w);
    *(uint2*)(xn + (size_t)row * 1024 + t * 4) = pk.u;
}

// ---------------------------------------------------------------------------
// Stage 1 GEMM: Y{re,im}^T[b][d][s] = (Xn @ {CD,SD})^T.  M=16384, N=1024, K=1024.
// Block tile 64M x 128N, 8 waves (2x4), wave tile 32x32 (2x2 WMMA tiles).
// Double-buffered LDS; async global->LDS staging when available.
// ---------------------------------------------------------------------------
__global__ __launch_bounds__(256) void gemm1(const _Float16* __restrict__ Xn,
                                             const _Float16* __restrict__ CD,
                                             const _Float16* __restrict__ SD,
                                             _Float16* __restrict__ YreT,
                                             _Float16* __restrict__ YimT) {
    __shared__ _Float16 As[2][64 * LDSS];
    __shared__ _Float16 Bc[2][128 * LDSS];
    __shared__ _Float16 Bs[2][128 * LDSS];

    const int t = threadIdx.x, lane = t & 31, wave = t >> 5;
    const int n0 = blockIdx.x * 128;                // over D
    const int m0 = blockIdx.y * 64;                 // over B*S
    const int wm = wave & 1, wn = wave >> 1;

    v8f accRe[2][2] = {};
    v8f accIm[2][2] = {};

    // Per-thread staging assignments (b128 granules).
    const int arow = t >> 2, achk = (t & 3) << 3;            // A: 64 rows x 4
    const int brow0 = t >> 2, bchk0 = (t & 3) << 3;          // B: 128 rows x 4,
    const int brow1 = (t + 256) >> 2, bchk1 = (t & 3) << 3;  //    two granules/thread

    const _Float16* gA  = Xn + (size_t)(m0 + arow) * 1024 + achk;
    const _Float16* gC0 = CD + (size_t)(n0 + brow0) * 1024 + bchk0;
    const _Float16* gC1 = CD + (size_t)(n0 + brow1) * 1024 + bchk1;
    const _Float16* gS0 = SD + (size_t)(n0 + brow0) * 1024 + bchk0;
    const _Float16* gS1 = SD + (size_t)(n0 + brow1) * 1024 + bchk1;

    auto stage = [&](int buf, int kk) {
        _Float16* lA  = &As[buf][arow * LDSS + achk];
        _Float16* lC0 = &Bc[buf][brow0 * LDSS + bchk0];
        _Float16* lC1 = &Bc[buf][brow1 * LDSS + bchk1];
        _Float16* lS0 = &Bs[buf][brow0 * LDSS + bchk0];
        _Float16* lS1 = &Bs[buf][brow1 * LDSS + bchk1];
#if USE_ASYNC
        async_b128(gA + kk, lA);
        async_b128(gC0 + kk, lC0);
        async_b128(gC1 + kk, lC1);
        async_b128(gS0 + kk, lS0);
        async_b128(gS1 + kk, lS1);
#else
        const uint4 r0 = *(const uint4*)(gA + kk);
        const uint4 r1 = *(const uint4*)(gC0 + kk);
        const uint4 r2 = *(const uint4*)(gC1 + kk);
        const uint4 r3 = *(const uint4*)(gS0 + kk);
        const uint4 r4 = *(const uint4*)(gS1 + kk);
        *(uint4*)lA = r0; *(uint4*)lC0 = r1; *(uint4*)lC1 = r2;
        *(uint4*)lS0 = r3; *(uint4*)lS1 = r4;
#endif
    };

    stage(0, 0);
#if USE_ASYNC
    wait_async0();
#endif
    __syncthreads();

    const int nsteps = 1024 / 32;
    for (int s = 0; s < nsteps; ++s) {
        const int cur = s & 1;
        if (s + 1 < nsteps) stage(cur ^ 1, (s + 1) * 32);

        v16h a[2], bc[2], bs[2];
#pragma unroll
        for (int i = 0; i < 2; ++i) {
            a[i]  = frag_a(&As[cur][0], wm * 32 + i * 16, lane);
            bc[i] = frag_b(&Bc[cur][0], wn * 32 + i * 16, lane);
            bs[i] = frag_b(&Bs[cur][0], wn * 32 + i * 16, lane);
        }
#pragma unroll
        for (int i = 0; i < 2; ++i)
#pragma unroll
            for (int j = 0; j < 2; ++j) {
                accRe[i][j] = __builtin_amdgcn_wmma_f32_16x16x32_f16(
                    false, a[i], false, bc[j], (short)0, accRe[i][j], false, false);
                accIm[i][j] = __builtin_amdgcn_wmma_f32_16x16x32_f16(
                    false, a[i], false, bs[j], (short)0, accIm[i][j], false, false);
            }
#if USE_ASYNC
        wait_async0();          // my async copies into the next buffer are done
#endif
        __syncthreads();        // everyone's copies visible; cur safe to reuse
    }

    // Transposed f16 store: lane holds fixed column n, 8 contiguous s-rows.
    const int b  = m0 >> 12;                        // batch (64 | 4096)
    const int sb = m0 & 4095;
    const int hi = lane >> 4, nl = lane & 15;
#pragma unroll
    for (int i = 0; i < 2; ++i)
#pragma unroll
        for (int j = 0; j < 2; ++j) {
            const int ncol = n0 + wn * 32 + j * 16 + nl;
            const int srow = sb + wm * 32 + i * 16 + hi * 8;
            const size_t base = ((size_t)(b * 1024 + ncol)) * 4096 + srow;
            PackU pr, pi;
#pragma unroll
            for (int r = 0; r < 8; ++r) {
                pr.h[r] = (_Float16)accRe[i][j][r];
                pi.h[r] = (_Float16)accIm[i][j][r];
            }
            *(uint4*)(YreT + base) = pr.u;
            *(uint4*)(YimT + base) = pi.u;
        }
}

// ---------------------------------------------------------------------------
// Stage 2 GEMM: Out[b] = CS @ Yre[b] + PS @ Yim[b].  Per batch M=4096, N=1024,
// K=4096.  Same tiling; B-operand is Y^T[b][n][k] from stage 1.
// ---------------------------------------------------------------------------
__global__ __launch_bounds__(256) void gemm2(const _Float16* __restrict__ CS,
                                             const _Float16* __restrict__ PS,
                                             const _Float16* __restrict__ YreT,
                                             const _Float16* __restrict__ YimT,
                                             float* __restrict__ out) {
    __shared__ _Float16 Ac[2][64 * LDSS];
    __shared__ _Float16 Ap[2][64 * LDSS];
    __shared__ _Float16 Br[2][128 * LDSS];
    __shared__ _Float16 Bi[2][128 * LDSS];

    const int t = threadIdx.x, lane = t & 31, wave = t >> 5;
    const int n0 = blockIdx.x * 128;                // over D
    const int m0 = blockIdx.y * 64;                 // over S
    const int b  = blockIdx.z;                      // batch
    const _Float16* Yre_b = YreT + (size_t)b * 1024 * 4096;
    const _Float16* Yim_b = YimT + (size_t)b * 1024 * 4096;
    const int wm = wave & 1, wn = wave >> 1;

    v8f acc[2][2] = {};

    const int arow = t >> 2, achk = (t & 3) << 3;
    const int brow0 = t >> 2, bchk0 = (t & 3) << 3;
    const int brow1 = (t + 256) >> 2, bchk1 = (t & 3) << 3;

    const _Float16* gAc = CS + (size_t)(m0 + arow) * 4096 + achk;
    const _Float16* gAp = PS + (size_t)(m0 + arow) * 4096 + achk;
    const _Float16* gR0 = Yre_b + (size_t)(n0 + brow0) * 4096 + bchk0;
    const _Float16* gR1 = Yre_b + (size_t)(n0 + brow1) * 4096 + bchk1;
    const _Float16* gI0 = Yim_b + (size_t)(n0 + brow0) * 4096 + bchk0;
    const _Float16* gI1 = Yim_b + (size_t)(n0 + brow1) * 4096 + bchk1;

    auto stage = [&](int buf, int kk) {
        _Float16* lAc = &Ac[buf][arow * LDSS + achk];
        _Float16* lAp = &Ap[buf][arow * LDSS + achk];
        _Float16* lR0 = &Br[buf][brow0 * LDSS + bchk0];
        _Float16* lR1 = &Br[buf][brow1 * LDSS + bchk1];
        _Float16* lI0 = &Bi[buf][brow0 * LDSS + bchk0];
        _Float16* lI1 = &Bi[buf][brow1 * LDSS + bchk1];
#if USE_ASYNC
        async_b128(gAc + kk, lAc);
        async_b128(gAp + kk, lAp);
        async_b128(gR0 + kk, lR0);
        async_b128(gR1 + kk, lR1);
        async_b128(gI0 + kk, lI0);
        async_b128(gI1 + kk, lI1);
#else
        const uint4 r0 = *(const uint4*)(gAc + kk);
        const uint4 r1 = *(const uint4*)(gAp + kk);
        const uint4 r2 = *(const uint4*)(gR0 + kk);
        const uint4 r3 = *(const uint4*)(gR1 + kk);
        const uint4 r4 = *(const uint4*)(gI0 + kk);
        const uint4 r5 = *(const uint4*)(gI1 + kk);
        *(uint4*)lAc = r0; *(uint4*)lAp = r1;
        *(uint4*)lR0 = r2; *(uint4*)lR1 = r3;
        *(uint4*)lI0 = r4; *(uint4*)lI1 = r5;
#endif
    };

    stage(0, 0);
#if USE_ASYNC
    wait_async0();
#endif
    __syncthreads();

    const int nsteps = 4096 / 32;
    for (int s = 0; s < nsteps; ++s) {
        const int cur = s & 1;
        if (s + 1 < nsteps) stage(cur ^ 1, (s + 1) * 32);

        v16h ac[2], ap[2], br[2], bi[2];
#pragma unroll
        for (int i = 0; i < 2; ++i) {
            ac[i] = frag_a(&Ac[cur][0], wm * 32 + i * 16, lane);
            ap[i] = frag_a(&Ap[cur][0], wm * 32 + i * 16, lane);
            br[i] = frag_b(&Br[cur][0], wn * 32 + i * 16, lane);
            bi[i] = frag_b(&Bi[cur][0], wn * 32 + i * 16, lane);
        }
#pragma unroll
        for (int i = 0; i < 2; ++i)
#pragma unroll
            for (int j = 0; j < 2; ++j) {
                acc[i][j] = __builtin_amdgcn_wmma_f32_16x16x32_f16(
                    false, ac[i], false, br[j], (short)0, acc[i][j], false, false);
                acc[i][j] = __builtin_amdgcn_wmma_f32_16x16x32_f16(
                    false, ap[i], false, bi[j], (short)0, acc[i][j], false, false);
            }
#if USE_ASYNC
        wait_async0();
#endif
        __syncthreads();
    }

    // f32 row-major store: Out[b][m][n]
    const int hi = lane >> 4, nl = lane & 15;
    float* outb = out + (size_t)b * 4096 * 1024;
#pragma unroll
    for (int i = 0; i < 2; ++i)
#pragma unroll
        for (int j = 0; j < 2; ++j) {
            const int mrow = m0 + wm * 32 + i * 16 + hi * 8;
            const int ncol = n0 + wn * 32 + j * 16 + nl;
#pragma unroll
            for (int r = 0; r < 8; ++r)
                outb[(size_t)(mrow + r) * 1024 + ncol] = acc[i][j][r];
        }
}

// ---------------------------------------------------------------------------
extern "C" void kernel_launch(void* const* d_in, const int* in_sizes, int n_in,
                              void* d_out, int out_size, void* d_ws, size_t ws_size,
                              hipStream_t stream) {
    const float* x     = (const float*)d_in[0];   // (4, 4096, 1024) f32
    const float* gamma = (const float*)d_in[1];   // (1024,) f32
    const float* beta  = (const float*)d_in[2];   // (1024,) f32
    float* out = (float*)d_out;                   // (4, 4096, 1024) f32

    char* ws = (char*)d_ws;
    size_t off = 0;
    auto take = [&](size_t bytes) -> char* {
        char* p = ws + off;
        off += (bytes + 255) & ~(size_t)255;
        return p;
    };
    _Float16* Xn   = (_Float16*)take((size_t)16384 * 1024 * 2); // LN output, f16
    _Float16* CD   = (_Float16*)take((size_t)1024 * 1024 * 2);  // cos, D
    _Float16* SD   = (_Float16*)take((size_t)1024 * 1024 * 2);  // -sin, D
    _Float16* CS   = (_Float16*)take((size_t)4096 * 4096 * 2);  // cos, S
    _Float16* PS   = (_Float16*)take((size_t)4096 * 4096 * 2);  // +sin, S
    _Float16* YreT = (_Float16*)take((size_t)4 * 1024 * 4096 * 2);
    _Float16* YimT = (_Float16*)take((size_t)4 * 1024 * 4096 * 2);
    if (off > ws_size) return;   // workspace too small: deterministic no-op

    fill_dft<<<(1024u * 1024u) / 256u, 256, 0, stream>>>(CD, SD, 10, -1.0f);
    fill_dft<<<(4096u * 4096u) / 256u, 256, 0, stream>>>(CS, PS, 12, 1.0f);
    ln_kernel<<<16384, 256, 0, stream>>>(x, gamma, beta, Xn);
    gemm1<<<dim3(8, 256), 256, 0, stream>>>(Xn, CD, SD, YreT, YimT);
    gemm2<<<dim3(8, 64, 4), 256, 0, stream>>>(CS, PS, YreT, YimT, out);
}
